// Attend_55448027791894
// MI455X (gfx1250) — matrix operand order repaired and checked by reading, not compiled, not run on hardware
//
#include <hip/hip_runtime.h>
#include <stdint.h>

// ---------------------------------------------------------------------------
// Flash-attention for the (buggy) reference: S = K·V^T * 0.125, causal mask,
// softmax, O = P·V.  q (d_in[0]) is unused by the reference.
// fp32 in/out; bf16 WMMA (v_wmma_f32_16x16x32_bf16) for both matmuls.
// CDNA5 async path: double-buffered GLOBAL_LOAD_ASYNC_TO_LDS_B128 prefetch of
// the next V tile (ASYNCcnt), overlapped with current-tile WMMA + softmax.
// Grid: (2048/64, B*H=64), 128 threads (4 wave32), each wave owns 16 rows.
// ---------------------------------------------------------------------------

typedef __attribute__((ext_vector_type(16))) __bf16 v16bf;
typedef __attribute__((ext_vector_type(8)))  __bf16 v8bf;
typedef __attribute__((ext_vector_type(4)))  __bf16 v4bf;
typedef __attribute__((ext_vector_type(8)))  float  v8f;

#define SEQ 2048
#define HDIM 64
#define BM 64            // query rows per workgroup
#define BN 64            // key/value cols per iteration
#define NWAVE 4
#define VPITCH 80        // bf16 elements per LDS row (32B-aligned runs, de-conflicted banks)

// Issue this thread's share (8 x b128 = 128B) of one 64x64 fp32 tile as async
// global->LDS DMA.  Per wave: 8 instructions -> ASYNCcnt +8.
__device__ __forceinline__ void async_tile_copy(const float* gbase, void* lds_base, int tid) {
  const char* g = (const char*)gbase;
  uint32_t l0 = (uint32_t)(uintptr_t)lds_base;
#pragma unroll
  for (int it = 0; it < 8; ++it) {
    int idx = tid + it * 128;                       // 1024 x 16B = 16KB tile
    uint64_t ga = (uint64_t)(uintptr_t)(g + (size_t)idx * 16);
    uint32_t la = l0 + (uint32_t)idx * 16;
    asm volatile("global_load_async_to_lds_b128 %0, %1, off"
                 :: "v"(la), "v"(ga) : "memory");
  }
}

__global__ __launch_bounds__(128) void attend_fa_kernel(
    const float* __restrict__ Kg, const float* __restrict__ Vg,
    float* __restrict__ Og) {

  __shared__ alignas(16) float  Vraw[2][BN][HDIM];        // async DMA staging (2 x 16KB)
  __shared__ alignas(32) __bf16 Vt [BN][VPITCH];          // V tile, row-major (j, d)
  __shared__ alignas(32) __bf16 VtT[HDIM][VPITCH];        // V tile, transposed (d, j)
  __shared__ alignas(32) __bf16 Pt [NWAVE][16][VPITCH];   // per-wave P staging (row, j)

  const int tid  = threadIdx.x;
  const int wave = tid >> 5;
  const int lane = tid & 31;
  const int lh   = lane >> 4;     // half-wave (selects K-chunk / row group)
  const int ln   = lane & 15;     // row (A) or column (B/C/D) index

  const int bh   = blockIdx.y;
  const int row0 = blockIdx.x * BM;
  const int m0   = row0 + wave * 16;           // this wave's first query row

  const float* Kb = Kg + (size_t)bh * SEQ * HDIM;
  const float* Vb = Vg + (size_t)bh * SEQ * HDIM;
  float*       Ob = Og + (size_t)bh * SEQ * HDIM;

  // ---- Load this wave's 16 K-rows into WMMA A layout (2 K-steps of 32) ----
  v16bf ak[2];
  {
    const float* kr = Kb + (size_t)(m0 + ln) * HDIM;
#pragma unroll
    for (int s = 0; s < 2; ++s) {
      const float4* pa = (const float4*)(kr + 32 * s + 16 * lh);       // K = base..base+7
      const float4* pb = (const float4*)(kr + 32 * s + 16 * lh + 8);   // K = base+8..+15
      float4 c0 = pa[0], c1 = pa[1], c2 = pb[0], c3 = pb[1];
      float t[16] = {c0.x,c0.y,c0.z,c0.w, c1.x,c1.y,c1.z,c1.w,
                     c2.x,c2.y,c2.z,c2.w, c3.x,c3.y,c3.z,c3.w};
#pragma unroll
      for (int i = 0; i < 16; ++i) ak[s][i] = (__bf16)t[i];
    }
  }

  // ---- Output accumulators + online-softmax state (8 rows/lane-half) ----
  v8f acc[4];
#pragma unroll
  for (int n = 0; n < 4; ++n)
#pragma unroll
    for (int r = 0; r < 8; ++r) acc[n][r] = 0.0f;
  float mrow[8], lrow[8];
#pragma unroll
  for (int r = 0; r < 8; ++r) { mrow[r] = -3.0e38f; lrow[r] = 0.0f; }

  const int nJT = blockIdx.x + 1;   // causal: only tiles with j0 <= row0+63

  // Prologue: async-prefetch tile 0 into buffer 0.
  async_tile_copy(Vb, &Vraw[0][0][0], tid);

  for (int jt = 0; jt < nJT; ++jt) {
    const int j0  = jt * BN;
    const int buf = jt & 1;

    // Prefetch next tile into the other buffer, then wait for current tile.
    if (jt + 1 < nJT) {
      async_tile_copy(Vb + (size_t)(j0 + BN) * HDIM, &Vraw[buf ^ 1][0][0], tid);
      asm volatile("s_wait_asynccnt 0x8" ::: "memory");   // current tile's 8 ops done
    } else {
      asm volatile("s_wait_asynccnt 0x0" ::: "memory");
    }
    __syncthreads();   // all waves' DMA landed; prev iteration's Vt/VtT readers done

    // ---- Convert staged fp32 tile -> bf16, row-major + transposed copies ----
    {
      const float4* src = (const float4*)&Vraw[buf][0][0];
#pragma unroll
      for (int it = 0; it < 8; ++it) {
        int idx = tid + it * 128;          // 1024 float4 = 64x64 floats
        int rr  = idx >> 4;                // row j (16 float4 per row)
        int cc  = (idx & 15) << 2;         // col d
        float4 f = src[idx];
        v4bf pk;
        pk[0] = (__bf16)f.x; pk[1] = (__bf16)f.y;
        pk[2] = (__bf16)f.z; pk[3] = (__bf16)f.w;
        *(v4bf*)&Vt[rr][cc] = pk;          // row-major copy
        VtT[cc + 0][rr] = pk[0];           // transposed copy
        VtT[cc + 1][rr] = pk[1];
        VtT[cc + 2][rr] = pk[2];
        VtT[cc + 3][rr] = pk[3];
      }
    }
    __syncthreads();

    // ---- S = K · V^T for this wave's 16 rows x 64 cols (4 N-tiles) ----
    v8f sc[4];
#pragma unroll
    for (int n = 0; n < 4; ++n) {
      v8f c;
#pragma unroll
      for (int r = 0; r < 8; ++r) c[r] = 0.0f;
#pragma unroll
      for (int s = 0; s < 2; ++s) {
        // B: col N = j0+16n+ln, K run = d in [32s+16*lh, +16) contiguous in Vt row
        v16bf b = *(const v16bf*)&Vt[n * 16 + ln][32 * s + 16 * lh];
        c = __builtin_amdgcn_wmma_f32_16x16x32_bf16(
                false, ak[s], false, b, (short)0, c, false, false);
      }
      sc[n] = c;
    }

    // ---- scale + causal mask ----
#pragma unroll
    for (int n = 0; n < 4; ++n) {
      const int jc = j0 + n * 16 + ln;
#pragma unroll
      for (int r = 0; r < 8; ++r) {
        const int ir = m0 + r + 8 * lh;
        float vv = sc[n][r] * 0.125f;
        sc[n][r] = (jc > ir) ? -3.0e38f : vv;
      }
    }

    // ---- online softmax (per row; reduce across 16 lanes of the half) ----
#pragma unroll
    for (int r = 0; r < 8; ++r) {
      float mx = fmaxf(fmaxf(sc[0][r], sc[1][r]), fmaxf(sc[2][r], sc[3][r]));
#pragma unroll
      for (int off = 1; off <= 8; off <<= 1)
        mx = fmaxf(mx, __shfl_xor(mx, off, 32));
      float mnew = fmaxf(mrow[r], mx);
      float fs   = __expf(mrow[r] - mnew);
      mrow[r]    = mnew;
      float rs = 0.0f;
#pragma unroll
      for (int n = 0; n < 4; ++n) {
        float p = __expf(sc[n][r] - mnew);
        sc[n][r] = p;
        rs += p;
        Pt[wave][r + 8 * lh][n * 16 + ln] = (__bf16)p;   // stage P (C-layout -> LDS)
      }
#pragma unroll
      for (int off = 1; off <= 8; off <<= 1)
        rs += __shfl_xor(rs, off, 32);
      lrow[r] = lrow[r] * fs + rs;
#pragma unroll
      for (int n = 0; n < 4; ++n) acc[n][r] *= fs;       // rescale O accumulator
    }
    __syncthreads();   // P staging visible / DS ordering

    // ---- O += P · V  (A = P from LDS in A-layout, B = VtT contiguous) ----
#pragma unroll
    for (int s = 0; s < 2; ++s) {
      v8bf lo = *(const v8bf*)&Pt[wave][ln][32 * s + 16 * lh];
      v8bf hi = *(const v8bf*)&Pt[wave][ln][32 * s + 16 * lh + 8];
      v16bf ap;
#pragma unroll
      for (int i = 0; i < 8; ++i) { ap[i] = lo[i]; ap[i + 8] = hi[i]; }
#pragma unroll
      for (int n = 0; n < 4; ++n) {
        // B: col N = d = 16n+ln, K run = j in [32s+16*lh, +16) contiguous in VtT row
        v16bf b = *(const v16bf*)&VtT[n * 16 + ln][32 * s + 16 * lh];
        acc[n] = __builtin_amdgcn_wmma_f32_16x16x32_bf16(
                     false, ap, false, b, (short)0, acc[n], false, false);
      }
    }
  }

  // ---- epilogue: divide by row sums, write fp32 output ----
#pragma unroll
  for (int r = 0; r < 8; ++r) {
    const float inv = 1.0f / lrow[r];
    float* orow = Ob + (size_t)(m0 + r + 8 * lh) * HDIM;
#pragma unroll
    for (int n = 0; n < 4; ++n)
      orow[n * 16 + ln] = acc[n][r] * inv;
  }
}

extern "C" void kernel_launch(void* const* d_in, const int* in_sizes, int n_in,
                              void* d_out, int out_size, void* d_ws, size_t ws_size,
                              hipStream_t stream) {
  (void)in_sizes; (void)n_in; (void)d_ws; (void)ws_size; (void)out_size;
  // Reference signature: (q, k, v) — q is genuinely unused by the reference.
  const float* k = (const float*)d_in[1];
  const float* v = (const float*)d_in[2];
  float* out = (float*)d_out;
  dim3 grid(SEQ / BM, 4 * 16);   // 32 row-blocks x (batch*heads)
  attend_fa_kernel<<<grid, 128, 0, stream>>>(k, v, out);
}
